// RouteOptimizerGCN_4569845203263
// MI455X (gfx1250) — matrix operand order repaired and checked by reading, not compiled, not run on hardware
//
#include <hip/hip_runtime.h>
#include <hip/hip_bf16.h>
#include <stdint.h>

// GCN (2 conv layers + linear head) for MI455X / gfx1250.
// Dense parts -> v_wmma_f32_16x16x32_f16 (f16 inputs, f32 accumulate).
// Sparse aggregation -> wave-per-edge L2-resident atomic scatter
// (h / agg tiles are 12.8 MB each, i.e. L2-resident in the 192 MB L2).

typedef __attribute__((ext_vector_type(16))) _Float16 v16h;
typedef __attribute__((ext_vector_type(8)))  _Float16 v8h;
typedef __attribute__((ext_vector_type(8)))  float    v8f;

#define FDIM 32   // hidden width of both GCN layers

// ---------------- degree / normalization ----------------

__global__ void k_init_deg(float* __restrict__ deg, int n) {
  int i = blockIdx.x * blockDim.x + threadIdx.x;
  if (i < n) deg[i] = 1.0f;               // self-loop contributes 1
}

__global__ void k_accum_deg(const long long* __restrict__ dst,
                            float* __restrict__ deg, long long e_count) {
  long long i = (long long)blockIdx.x * blockDim.x + threadIdx.x;
  if (i < e_count) atomicAdd(&deg[dst[i]], 1.0f);
}

__global__ void k_dinv(float* __restrict__ deg, int n) {
  int i = blockIdx.x * blockDim.x + threadIdx.x;
  if (i < n) deg[i] = rsqrtf(deg[i]);     // deg >= 1 always (self loops)
}

// ---------------- f32 -> f16 conversions ----------------

__global__ void k_cvt_f16(const float* __restrict__ in,
                          _Float16* __restrict__ out, long long n) {
  long long i = (long long)blockIdx.x * blockDim.x + threadIdx.x;
  if (i < n) out[i] = (_Float16)in[i];
}

__global__ void k_relu_cvt_f16(const float* __restrict__ in,
                               _Float16* __restrict__ out, long long n) {
  long long i = (long long)blockIdx.x * blockDim.x + threadIdx.x;
  if (i < n) out[i] = (_Float16)fmaxf(in[i], 0.0f);
}

// ---------------- WMMA GEMM: C[M,N] = A[M,K] * B[K,N] ----------------
// A, B f16 row-major; C f32 row-major. One wave32 per 16x16 C tile.
// A fragment layout (ISA 7.12.2, 16-bit A 16x32): lane (l&15) = row,
// half-wave (l>>4) selects K chunks [g*8 .. g*8+7] and [16+g*8 .. 16+g*8+7].
// B fragment: lane (l&15) = column, same K chunks as A per half-wave.
template <int K, int N>
__global__ void k_gemm_wmma(const _Float16* __restrict__ A,
                            const _Float16* __restrict__ B,
                            float* __restrict__ C, int M) {
  const int lane = threadIdx.x & 31;
  const int wave = threadIdx.x >> 5;
  const int wpb  = blockDim.x >> 5;
  const int ntile = N / 16;
  int tile = blockIdx.x * wpb + wave;
  int mt = tile / ntile;
  int nt = tile - mt * ntile;
  int m0 = mt * 16;
  if (m0 >= M) return;

  const int grp = lane >> 4;      // 0 or 1
  const int mr  = lane & 15;      // row within tile

  v8f acc = {};
#pragma unroll
  for (int kt = 0; kt < K; kt += 32) {
    const _Float16* arow = A + (size_t)(m0 + mr) * K + kt + grp * 8;
    v8h alo = *(const v8h*)(arow);        // K = kt + grp*8 + [0..7]
    v8h ahi = *(const v8h*)(arow + 16);   // K = kt + 16 + grp*8 + [0..7]
    v16h a;
#pragma unroll
    for (int i = 0; i < 8; ++i) { a[i] = alo[i]; a[i + 8] = ahi[i]; }

    v16h b;
#pragma unroll
    for (int i = 0; i < 8; ++i) {
      b[i]     = B[(size_t)(kt + grp * 8 + i) * N + nt * 16 + mr];
      b[i + 8] = B[(size_t)(kt + 16 + grp * 8 + i) * N + nt * 16 + mr];
    }

    acc = __builtin_amdgcn_wmma_f32_16x16x32_f16(
        /*neg_a=*/false, a, /*neg_b=*/false, b,
        /*c_mod=*/(short)0, acc, /*reuse_a=*/false, /*reuse_b=*/false);
  }

  // C layout: VGPR r, lanes 0-15 -> M=r, lanes 16-31 -> M=8+r; col = lane&15.
#pragma unroll
  for (int r = 0; r < 8; ++r)
    C[(size_t)(m0 + grp * 8 + r) * N + nt * 16 + mr] = acc[r];
}

// ---------------- aggregation ----------------

// agg[i][j] = bias[j] + h[i][j] * dinv[i]^2   (bias + self-loop message)
__global__ void k_init_agg(const float* __restrict__ h,
                           const float* __restrict__ dinv,
                           const float* __restrict__ bias,
                           float* __restrict__ agg, int n) {
  long long idx = (long long)blockIdx.x * blockDim.x + threadIdx.x;
  if (idx < (long long)n * FDIM) {
    int i = (int)(idx >> 5);      // FDIM == 32
    int j = (int)(idx & 31);
    float di = dinv[i];
    agg[idx] = bias[j] + h[idx] * di * di;
  }
}

// One wave32 per edge; lane j handles feature j. Gather h[src], scale by
// dinv[src]*dinv[dst], atomic scatter-add into agg[dst]. All L2-resident.
__global__ void k_edge_agg(const long long* __restrict__ src,
                           const long long* __restrict__ dst,
                           const float* __restrict__ dinv,
                           const float* __restrict__ h,
                           float* __restrict__ agg, long long e_count) {
  long long e = (long long)blockIdx.x * (blockDim.x >> 5) + (threadIdx.x >> 5);
  if (e >= e_count) return;
  int j = threadIdx.x & 31;
  long long s = src[e];
  long long d = dst[e];
  float norm = dinv[s] * dinv[d];
  float v = h[s * FDIM + j] * norm;
  atomicAdd(&agg[d * FDIM + j], v);
}

// scores[i] = sum_j relu(agg[i][j]) * Wfc[j] + bfc
__global__ void k_final(const float* __restrict__ agg,
                        const float* __restrict__ wfc,
                        const float* __restrict__ bfc,
                        float* __restrict__ out, int n) {
  int i = blockIdx.x * blockDim.x + threadIdx.x;
  if (i >= n) return;
  float s = bfc[0];
#pragma unroll
  for (int j = 0; j < FDIM; ++j)
    s += fmaxf(agg[(size_t)i * FDIM + j], 0.0f) * wfc[j];
  out[i] = s;
}

// ---------------- launcher ----------------

extern "C" void kernel_launch(void* const* d_in, const int* in_sizes, int n_in,
                              void* d_out, int out_size, void* d_ws, size_t ws_size,
                              hipStream_t stream) {
  const float*     x   = (const float*)d_in[0];
  const long long* ei  = (const long long*)d_in[1];   // int64 [2, E]
  const float*     W1  = (const float*)d_in[2];
  const float*     b1  = (const float*)d_in[3];
  const float*     W2  = (const float*)d_in[4];
  const float*     b2  = (const float*)d_in[5];
  const float*     Wfc = (const float*)d_in[6];
  const float*     bfc = (const float*)d_in[7];
  float*           out = (float*)d_out;

  const int IN_F = 128;
  const int n = in_sizes[0] / IN_F;                   // 100000 (divisible by 16)
  const long long E = in_sizes[1] / 2;                // 3 200 000
  const long long* src = ei;
  const long long* dst = ei + E;

  // carve workspace (256 B aligned slices)
  char* p = (char*)d_ws;
  auto carve = [&](size_t bytes) -> char* {
    char* r = p;
    p += (bytes + 255) & ~(size_t)255;
    return r;
  };
  float*    dinv = (float*)carve(sizeof(float) * (size_t)n);
  _Float16* xh   = (_Float16*)carve(sizeof(_Float16) * (size_t)n * IN_F);
  _Float16* W1h  = (_Float16*)carve(sizeof(_Float16) * IN_F * FDIM);
  _Float16* W2h  = (_Float16*)carve(sizeof(_Float16) * FDIM * FDIM);
  float*    h1   = (float*)carve(sizeof(float) * (size_t)n * FDIM);    // reused as h2
  float*    agg  = (float*)carve(sizeof(float) * (size_t)n * FDIM);    // reused layer 2
  _Float16* h2h  = (_Float16*)carve(sizeof(_Float16) * (size_t)n * FDIM);
  (void)ws_size; (void)n_in; (void)out_size;

  const long long nF = (long long)n * FDIM;
  const long long nX = (long long)n * IN_F;

  // 1) symmetric-norm degrees (self-loops folded into init value)
  k_init_deg<<<(n + 255) / 256, 256, 0, stream>>>(dinv, n);
  k_accum_deg<<<(int)((E + 255) / 256), 256, 0, stream>>>(dst, dinv, E);
  k_dinv<<<(n + 255) / 256, 256, 0, stream>>>(dinv, n);

  // 2) f16 copies of GEMM inputs
  k_cvt_f16<<<(int)((nX + 255) / 256), 256, 0, stream>>>(x, xh, nX);
  k_cvt_f16<<<(IN_F * FDIM + 255) / 256, 256, 0, stream>>>(W1, W1h, IN_F * FDIM);
  k_cvt_f16<<<(FDIM * FDIM + 255) / 256, 256, 0, stream>>>(W2, W2h, FDIM * FDIM);

  // 3) layer 1: h1 = x @ W1 via WMMA (tiles: 6250 M-tiles x 2 N-tiles)
  {
    int tiles = (n / 16) * (FDIM / 16);
    int wpb = 4;  // 128 threads = 4 waves per block
    k_gemm_wmma<128, FDIM><<<(tiles + wpb - 1) / wpb, wpb * 32, 0, stream>>>(
        xh, W1h, h1, n);
  }
  k_init_agg<<<(int)((nF + 255) / 256), 256, 0, stream>>>(h1, dinv, b1, agg, n);
  k_edge_agg<<<(int)((E + 7) / 8), 256, 0, stream>>>(src, dst, dinv, h1, agg, E);

  // 4) relu + cvt -> GEMM2 input
  k_relu_cvt_f16<<<(int)((nF + 255) / 256), 256, 0, stream>>>(agg, h2h, nF);

  // 5) layer 2: h2 = relu(h1_agg) @ W2 via WMMA (K=32, single WMMA per tile)
  {
    int tiles = (n / 16) * (FDIM / 16);
    int wpb = 4;
    k_gemm_wmma<FDIM, FDIM><<<(tiles + wpb - 1) / wpb, wpb * 32, 0, stream>>>(
        h2h, W2h, h1, n);   // h1 buffer reused as h2
  }
  k_init_agg<<<(int)((nF + 255) / 256), 256, 0, stream>>>(h1, dinv, b2, agg, n);
  k_edge_agg<<<(int)((E + 7) / 8), 256, 0, stream>>>(src, dst, dinv, h1, agg, E);

  // 6) head: scores = relu(agg) @ Wfc + bfc
  k_final<<<(n + 255) / 256, 256, 0, stream>>>(agg, Wfc, bfc, out, n);
}
